// MultiHeadAttention_91018946937248
// MI455X (gfx1250) — matrix operand order repaired and checked by reading, not compiled, not run on hardware
//
#include <hip/hip_runtime.h>

#define SEQ     2048
#define BATCH   2
#define DMODEL  1024
#define NHEADS  16
#define HEADDIM 64

typedef __attribute__((ext_vector_type(16))) __bf16 v16bf;
typedef __attribute__((ext_vector_type(8)))  __bf16 v8bf;
typedef __attribute__((ext_vector_type(8)))  float  v8f;

// ---------- bf16 helpers (round-to-nearest-even) ----------
static __device__ __forceinline__ unsigned short bf_bits(float f) {
    unsigned u = __builtin_bit_cast(unsigned, f);
    u += 0x7FFFu + ((u >> 16) & 1u);
    return (unsigned short)(u >> 16);
}
static __device__ __forceinline__ unsigned pack2bf(float lo, float hi) {
    return (unsigned)bf_bits(lo) | ((unsigned)bf_bits(hi) << 16);
}

// ---------- fragment layout used here (wave32, bf16 16x16x32; contiguous k, not ISA table order) ----------
// A: lane holds row M=lane%16; elements {0..7} = K[2*baseA..2*baseA+7],
//    elements {8..15} = K[2*baseA+8..2*baseA+15], baseA = (lane<16)?0:8.
// B: lane holds col N=lane%16; elements i = K[baseB + i], baseB = (lane<16)?0:16.
// C/D: lane holds col N=lane%16; vgpr v = row M = v + ((lane<16)?0:8).
static __device__ __forceinline__ v16bf load_fragA(const __bf16* p0, const __bf16* p1) {
    v8bf lo = *(const v8bf*)p0;
    v8bf hi = *(const v8bf*)p1;
    return __builtin_shufflevector(lo, hi, 0,1,2,3,4,5,6,7,8,9,10,11,12,13,14,15);
}
static __device__ __forceinline__ v8f wmma_bf16(v16bf a, v16bf b, v8f c) {
    return __builtin_amdgcn_wmma_f32_16x16x32_bf16(false, a, false, b, (short)0, c, false, false);
}

// ============================================================================
// Pre-pass A: x (f32 [s,b,d]) -> xb (bf16 [b*S+s, d]). Coalesced both sides.
// ============================================================================
__global__ void convert_x_kernel(const float* __restrict__ x,
                                 unsigned* __restrict__ xb_u32) {
    int gid = blockIdx.x * 256 + threadIdx.x;        // 2M threads, 2 elems each
    int m = gid >> 9;                                 // b*S+s
    int dp = gid & 511;                               // d/2
    int b = m >> 11, s = m & (SEQ - 1);
    const float2 v = *(const float2*)&x[((size_t)(s * BATCH + b)) * DMODEL + 2 * dp];
    xb_u32[(size_t)m * 512 + dp] = pack2bf(v.x, v.y);
}

// ============================================================================
// Pre-pass B: weights f32 [K=1024, N] -> bf16 packed in B-fragment tile order:
// dst[(nt*32 + kt)*512 + lane*16 + i], one wave per 32x16 tile.
// ============================================================================
__global__ void pack_w_kernel(const float* __restrict__ src,
                              unsigned short* __restrict__ dst, int N) {
    const int lane = threadIdx.x & 31;
    const int wave = threadIdx.x >> 5;
    const int t = blockIdx.x * 8 + wave;             // nt*32 + kt
    const int nt = t >> 5, kt = t & 31;
    const int n = nt * 16 + (lane & 15);
    const int kbase = kt * 32 + ((lane < 16) ? 0 : 16);
    unsigned short* out = dst + (size_t)t * 512 + lane * 16;
#pragma unroll
    for (int i = 0; i < 16; ++i)
        out[i] = bf_bits(src[(size_t)(kbase + i) * N + n]);
}

// ============================================================================
// Kernel 1: QKV projection (M=4096, N=3072, K=1024) + bias + RoPE.
// qbuf/kbuf: [b,h,s,dh]; vTbuf: [b,h,dh,s]  (all bf16)
// ============================================================================
__global__ void qkv_rope_kernel(const __bf16* __restrict__ xb,
                                const unsigned short* __restrict__ WqkvP,
                                const float* __restrict__ bqkv,
                                unsigned short* __restrict__ qbuf,
                                unsigned short* __restrict__ kbuf,
                                unsigned short* __restrict__ vTbuf) {
    const int lane = threadIdx.x & 31;
    const int wave = threadIdx.x >> 5;
    const int tile = blockIdx.x * 8 + wave;          // 256*192 tiles
    const int NT   = 3 * DMODEL / 16;                // 192
    const int mt   = tile / NT, nt = tile % NT;
    const int ln   = lane & 15;
    const int baseA = (lane < 16) ? 0 : 8;
    const int mA   = mt * 16 + ln;                   // A row (= b*S+s)
    const int n    = nt * 16 + ln;                   // B/D column
    const __bf16* arow = xb + (size_t)mA * DMODEL;
    const __bf16* bsrc = (const __bf16*)(WqkvP) + ((size_t)nt * 32) * 512 + lane * 16;

    v8f acc = {0.f,0.f,0.f,0.f,0.f,0.f,0.f,0.f};
    for (int k0 = 0; k0 < DMODEL; k0 += 32) {
        __builtin_prefetch(bsrc + 512, 0, 1);
        v16bf a = load_fragA(arow + k0 + 2 * baseA, arow + k0 + 2 * baseA + 8);
        v16bf bw = *(const v16bf*)bsrc;
        bsrc += 512;
        acc = wmma_bf16(a, bw, acc);
    }

    const float bias = bqkv[n];
    const int which  = n / DMODEL;                   // 0=q 1=k 2=v (wave-uniform)
    const int within = n % DMODEL;
    const int h = within >> 6, dh = within & 63;
    const float invf = __expf(-(float)(dh & ~1) * (9.210340372f / 64.0f)); // base^(-2j/64)

#pragma unroll
    for (int v = 0; v < 8; ++v) {
        int mr = mt * 16 + v + baseA;
        int bb = mr >> 11, ss = mr & (SEQ - 1);
        float val = acc[v] + bias;
        if (which == 2) {
            vTbuf[(((size_t)bb * NHEADS + h) * HEADDIM + dh) * SEQ + ss] = bf_bits(val);
        } else {
            float partner = __shfl_xor(val, 1, 32);           // column n^1, same row
            float rot = (dh & 1) ? partner : -partner;        // rotate_every_two
            float ang = (float)ss * invf;
            float r = val * __cosf(ang) + rot * __sinf(ang);
            unsigned short* dst = (which == 0) ? qbuf : kbuf;
            dst[(((size_t)bb * NHEADS + h) * SEQ + ss) * HEADDIM + dh] = bf_bits(r);
        }
    }
}

// ============================================================================
// Kernel 2: causal flash attention. One wave = (b,h) x 16-query tile.
// attnbuf: bf16 [b*S+s, h*64+dh]  (row-major [4096, 1024])
// ============================================================================
__global__ void attn_kernel(const unsigned short* __restrict__ qbuf,
                            const unsigned short* __restrict__ kbuf,
                            const unsigned short* __restrict__ vTbuf,
                            unsigned short* __restrict__ attnbuf) {
    __shared__ unsigned short Plds[8][16 * 32];
    const int lane = threadIdx.x & 31;
    const int wave = threadIdx.x >> 5;
    const int gw   = blockIdx.x * 8 + wave;          // 4096 waves
    const int qt   = gw & (SEQ / 16 - 1);            // 128 q tiles
    const int bh   = gw >> 7;                        // 32 (b*16+h)
    const int ln   = lane & 15;
    const int baseA = (lane < 16) ? 0 : 8;
    const int baseB = (lane < 16) ? 0 : 16;

    const __bf16* Qh  = (const __bf16*)(qbuf  + (size_t)bh * SEQ * HEADDIM);
    const __bf16* Kh  = (const __bf16*)(kbuf  + (size_t)bh * SEQ * HEADDIM);
    const __bf16* VTh = (const __bf16*)(vTbuf + (size_t)bh * HEADDIM * SEQ);
    const __bf16* Pw  = (const __bf16*)Plds[wave];

    // Q A-fragments: dh halves [0,32) and [32,64)
    const __bf16* qrowp = Qh + (size_t)(qt * 16 + ln) * HEADDIM;
    v16bf qa0 = load_fragA(qrowp + 2 * baseA,      qrowp + 2 * baseA + 8);
    v16bf qa1 = load_fragA(qrowp + 32 + 2 * baseA, qrowp + 40 + 2 * baseA);

    float rmax[8], rsum[8];
    v8f o[4];
#pragma unroll
    for (int v = 0; v < 8; ++v) { rmax[v] = -__builtin_inff(); rsum[v] = 0.f; }
#pragma unroll
    for (int t = 0; t < 4; ++t) o[t] = (v8f){0.f,0.f,0.f,0.f,0.f,0.f,0.f,0.f};

    const float scale = 0.125f;                      // 64^-0.5
    const int qlast = qt * 16 + 15;

    for (int kc = 0; kc <= qlast; kc += 32) {
        // ---- scores for 32 keys (two 16-key subtiles) ----
        v8f sc[2];
#pragma unroll
        for (int t = 0; t < 2; ++t) {
            const __bf16* krow = Kh + (size_t)(kc + t * 16 + ln) * HEADDIM;
            v16bf b0 = *(const v16bf*)(krow + baseB);
            v16bf b1 = *(const v16bf*)(krow + 32 + baseB);
            v8f c = {0.f,0.f,0.f,0.f,0.f,0.f,0.f,0.f};
            c = wmma_bf16(qa0, b0, c);
            c = wmma_bf16(qa1, b1, c);
            sc[t] = c;
        }
        // ---- causal mask + online softmax ----
#pragma unroll
        for (int v = 0; v < 8; ++v) {
            int qr = qt * 16 + v + baseA;
            float s0 = (kc + ln      <= qr) ? sc[0][v] * scale : -__builtin_inff();
            float s1 = (kc + 16 + ln <= qr) ? sc[1][v] * scale : -__builtin_inff();
            float mx = fmaxf(s0, s1);
#pragma unroll
            for (int off = 1; off < 16; off <<= 1) mx = fmaxf(mx, __shfl_xor(mx, off, 16));
            float nmax = fmaxf(rmax[v], mx);
            float corr = __expf(rmax[v] - nmax);
            float p0 = __expf(s0 - nmax);
            float p1 = __expf(s1 - nmax);
            float ps = p0 + p1;
#pragma unroll
            for (int off = 1; off < 16; off <<= 1) ps += __shfl_xor(ps, off, 16);
            rsum[v] = rsum[v] * corr + ps;
            rmax[v] = nmax;
#pragma unroll
            for (int t = 0; t < 4; ++t) o[t][v] *= corr;
            int row = v + baseA;
            Plds[wave][row * 32 + ln]      = bf_bits(p0);
            Plds[wave][row * 32 + 16 + ln] = bf_bits(p1);
        }
        asm volatile("s_wait_dscnt 0" ::: "memory");
        // ---- re-layout P (D-frag -> A-frag) via LDS: two ds_load_b128 ----
        v16bf pa = load_fragA(Pw + ln * 32 + 2 * baseA, Pw + ln * 32 + 2 * baseA + 8);
        // ---- O += P * V (dh in four 16-col subtiles) ----
#pragma unroll
        for (int t = 0; t < 4; ++t) {
            const __bf16* vrow = VTh + (size_t)(t * 16 + ln) * SEQ + kc;
            v16bf vb = *(const v16bf*)(vrow + baseB);
            o[t] = wmma_bf16(pa, vb, o[t]);
        }
    }

    // ---- normalize + store attn (bf16, [b*S+s, 1024]) ----
    const int h = bh & (NHEADS - 1), b = bh >> 4;
#pragma unroll
    for (int v = 0; v < 8; ++v) {
        int qr = qt * 16 + v + baseA;
        float inv = 1.0f / rsum[v];
#pragma unroll
        for (int t = 0; t < 4; ++t) {
            attnbuf[((size_t)(b * SEQ + qr)) * DMODEL + h * HEADDIM + t * 16 + ln] =
                bf_bits(o[t][v] * inv);
        }
    }
}

// ============================================================================
// Kernel 3: output projection (M=4096, N=1024, K=1024) + bias -> out[s,b,d] f32
// ============================================================================
__global__ void out_proj_kernel(const unsigned short* __restrict__ attnbuf,
                                const unsigned short* __restrict__ WoutP,
                                const float* __restrict__ bout,
                                float* __restrict__ out) {
    const int lane = threadIdx.x & 31;
    const int wave = threadIdx.x >> 5;
    const int tile = blockIdx.x * 8 + wave;          // 256*64 tiles
    const int NT = DMODEL / 16;                      // 64
    const int mt = tile / NT, nt = tile % NT;
    const int ln = lane & 15;
    const int baseA = (lane < 16) ? 0 : 8;
    const int m  = mt * 16 + ln;                     // b*S+s
    const int n  = nt * 16 + ln;
    const __bf16* arow = (const __bf16*)attnbuf + (size_t)m * DMODEL;
    const __bf16* bsrc = (const __bf16*)(WoutP) + ((size_t)nt * 32) * 512 + lane * 16;

    v8f acc = {0.f,0.f,0.f,0.f,0.f,0.f,0.f,0.f};
    for (int k0 = 0; k0 < DMODEL; k0 += 32) {
        __builtin_prefetch(bsrc + 512, 0, 1);
        v16bf a = load_fragA(arow + k0 + 2 * baseA, arow + k0 + 2 * baseA + 8);
        v16bf bw = *(const v16bf*)bsrc;
        bsrc += 512;
        acc = wmma_bf16(a, bw, acc);
    }
    const float bias = bout[n];
#pragma unroll
    for (int v = 0; v < 8; ++v) {
        int mr = mt * 16 + v + baseA;
        int b = mr >> 11, s = mr & (SEQ - 1);
        out[((size_t)s * BATCH + b) * DMODEL + n] = acc[v] + bias;
    }
}

// ============================================================================
extern "C" void kernel_launch(void* const* d_in, const int* in_sizes, int n_in,
                              void* d_out, int out_size, void* d_ws, size_t ws_size,
                              hipStream_t stream) {
    const float* x    = (const float*)d_in[0];
    const float* Wqkv = (const float*)d_in[1];
    const float* bqkv = (const float*)d_in[2];
    const float* Wout = (const float*)d_in[3];
    const float* bout = (const float*)d_in[4];
    float* out = (float*)d_out;

    char* ws = (char*)d_ws;
    const size_t MB = (size_t)1024 * 1024;
    unsigned short* qbuf    = (unsigned short*)(ws);            //  8 MB
    unsigned short* kbuf    = (unsigned short*)(ws +  8 * MB);  //  8 MB
    unsigned short* vTbuf   = (unsigned short*)(ws + 16 * MB);  //  8 MB
    unsigned short* attnbuf = (unsigned short*)(ws + 24 * MB);  //  8 MB
    unsigned short* xb      = (unsigned short*)(ws + 32 * MB);  //  8 MB
    unsigned short* WqkvP   = (unsigned short*)(ws + 40 * MB);  //  6 MB
    unsigned short* WoutP   = (unsigned short*)(ws + 46 * MB);  //  2 MB

    // pre-conversion / packing (memory-bound, one pass)
    convert_x_kernel<<<8192, 256, 0, stream>>>(x, (unsigned*)xb);
    pack_w_kernel<<<768, 256, 0, stream>>>(Wqkv, WqkvP, 3 * DMODEL);  // 6144 tiles
    pack_w_kernel<<<256, 256, 0, stream>>>(Wout, WoutP, DMODEL);      // 2048 tiles

    // 49152 tiles / 8 waves per block
    qkv_rope_kernel<<<6144, 256, 0, stream>>>((const __bf16*)xb, WqkvP, bqkv,
                                              qbuf, kbuf, vTbuf);
    // 4096 waves / 8
    attn_kernel<<<512, 256, 0, stream>>>(qbuf, kbuf, vTbuf, attnbuf);
    // 16384 tiles / 8
    out_proj_kernel<<<2048, 256, 0, stream>>>(attnbuf, WoutP, bout, out);
}
